// Attention_34574486733032
// MI455X (gfx1250) — matrix operand order repaired and checked by reading, not compiled, not run on hardware
//
#include <hip/hip_runtime.h>
#include <hip/hip_bf16.h>
#include <stdint.h>

// ---------------------------------------------------------------------------
// MI455X (gfx1250) attention, bf16 WMMA (v_wmma_f32_16x16x32_bf16), wave32.
// Pipelined attention loop + async global->LDS staging; spill-free projections.
// ---------------------------------------------------------------------------

typedef __attribute__((ext_vector_type(16))) __bf16 v16bf;
typedef __attribute__((ext_vector_type(8)))  __bf16 v8bf;
typedef __attribute__((ext_vector_type(8)))  float  v8f;

union Frag16 { v16bf v; v8bf h[2]; };

#define DEV __device__ __forceinline__

// Problem constants
#define BSZ 4
#define SEQ 2048
#define DIM 128
#define NH  8
#define HD  16
#define ROWS (BSZ*SEQ)     // 8192
#define NC  (SEQ/32)       // 64 key chunks of 32

#if defined(__has_builtin)
# if __has_builtin(__builtin_amdgcn_global_load_async_to_lds_b128) && \
     __has_builtin(__builtin_amdgcn_s_wait_asynccnt)
#  define USE_ASYNC_LDS 1
# endif
#endif

DEV v8bf zero8() {
  v8bf z;
#pragma unroll
  for (int i = 0; i < 8; ++i) z[i] = (__bf16)0.0f;
  return z;
}

// A-fragment (16x32 bf16, row-major, leading dim ld):
// lane: row = lane&15, half = lane>>4
// elems 0..15 : K = half*16 + 0..15 (contig)
DEV v16bf loadA(const __bf16* base, int ld) {
  const int lane = threadIdx.x & 31;
  const int row = lane & 15, half = lane >> 4;
  const __bf16* p = base + row * ld + half * 16;
  Frag16 f;
  f.h[0] = *(const v8bf*)(p);
  f.h[1] = *(const v8bf*)(p + 8);
  return f.v;
}

// A-fragment with K padded 16->32 (K >= 16 zero): hd=16 contraction.
DEV v16bf loadA_pad16(const __bf16* base, int ld) {
  const int lane = threadIdx.x & 31;
  const int row = lane & 15, half = lane >> 4;
  Frag16 f;
  f.h[0] = half ? zero8() : *(const v8bf*)(base + row * ld);
  f.h[1] = half ? zero8() : *(const v8bf*)(base + row * ld + 8);
  return f.v;
}

// B-fragment (32x16) from transposed storage Bt[N][K] row-major:
// lane: n = lane&15, half = lane>>4; elems 0..15 : K = half*16 + 0..15 (contig)
DEV v16bf loadB(const __bf16* bt, int ld) {
  const int lane = threadIdx.x & 31;
  const int n = lane & 15, half = lane >> 4;
  const __bf16* p = bt + n * ld + half * 16;
  Frag16 f;
  f.h[0] = *(const v8bf*)(p);
  f.h[1] = *(const v8bf*)(p + 8);
  return f.v;
}

// B-fragment with K padded 16->32 (upper half-wave lanes carry zeros).
DEV v16bf loadB_pad16(const __bf16* bt, int ld) {
  const int lane = threadIdx.x & 31;
  const int n = lane & 15, half = lane >> 4;
  Frag16 f;
  if (half == 0) {
    const __bf16* p = bt + n * ld;
    f.h[0] = *(const v8bf*)(p);
    f.h[1] = *(const v8bf*)(p + 8);
  } else {
    f.h[0] = zero8();
    f.h[1] = zero8();
  }
  return f.v;
}

DEV v8f wmma_bf16(v16bf a, v16bf b, v8f c) {
  return __builtin_amdgcn_wmma_f32_16x16x32_bf16(
      false, a, false, b, (short)0, c, false, false);
}

#if defined(USE_ASYNC_LDS)
// Exact pointee type per hipcc diagnostic: GCC vector int32x4 (vector_size 16),
// AS1 (global) for src, AS3 (LDS) for dst.
typedef int i32x4 __attribute__((vector_size(16)));
typedef __attribute__((address_space(1))) i32x4* as1_i32x4_ptr;
typedef __attribute__((address_space(3))) i32x4* as3_i32x4_ptr;

// Per-lane 16B async copy, global -> LDS (ASYNCcnt-tracked).
// Generic LDS address low 32 bits == LDS byte offset (aperture layout, ISA 10.2).
DEV void async_copy16(const void* g, void* l) {
  __builtin_amdgcn_global_load_async_to_lds_b128(
      (as1_i32x4_ptr)(uintptr_t)g,
      (as3_i32x4_ptr)(uint32_t)(uintptr_t)l,
      0, 0);
}
// Stage one 32x16 V tile (Bt rows = vT) into LDS: 2 async ops per lane.
DEV void stage_v(const __bf16* vchunk, __bf16* lbuf) {
  const int lane = threadIdx.x & 31;
  const int n = lane & 15, half = lane >> 4;
  const __bf16* g = vchunk + n * SEQ + half * 16;  // lane's own 32B slice
  __bf16* l = lbuf + lane * 16;
  async_copy16(g, l);
  async_copy16(g + 8, l + 8);
}
// Read the staged B-fragment back (each lane reads its own slice).
DEV v16bf ldsB(const __bf16* lbuf) {
  const int lane = threadIdx.x & 31;
  Frag16 f;
  f.h[0] = *(const v8bf*)(lbuf + lane * 16);
  f.h[1] = *(const v8bf*)(lbuf + lane * 16 + 8);
  return f.v;
}
#endif

// ---------------------------------------------------------------------------
// f32 -> bf16 conversion (+ transpose of QKV weights into Bt layout)
// ---------------------------------------------------------------------------
__global__ void cvt_x_kernel(const float* __restrict__ x, __bf16* __restrict__ xb, int n) {
  int i = blockIdx.x * blockDim.x + threadIdx.x;
  if (i < n) xb[i] = (__bf16)x[i];
}

__global__ void cvt_w_kernel(const float* __restrict__ wq, const float* __restrict__ wk,
                             const float* __restrict__ wv, const float* __restrict__ lw,
                             __bf16* __restrict__ wqT, __bf16* __restrict__ wkT,
                             __bf16* __restrict__ wvT, __bf16* __restrict__ linT) {
  int i = blockIdx.x * blockDim.x + threadIdx.x;
  if (i < DIM * DIM) {
    int k = i / DIM, n = i % DIM;
    wqT[n * DIM + k] = (__bf16)wq[i];   // x @ W   -> Bt[n][k] = W[k][n]
    wkT[n * DIM + k] = (__bf16)wk[i];
    wvT[n * DIM + k] = (__bf16)wv[i];
    linT[i] = (__bf16)lw[i];            // a @ W^T -> Bt[n][k] = lin_w[n][k]
  }
}

// ---------------------------------------------------------------------------
// QKV projection: one wave per 16-row tile; 24 output tiles, 96 WMMAs.
// Per-tile: 8 clustered B loads -> one wait -> 4 back-to-back WMMAs.
// No cross-tile buffering (occupancy hides latency; keeps VGPRs spill-free).
// ---------------------------------------------------------------------------
__global__ void qkv_kernel(const __bf16* __restrict__ xb,
                           const __bf16* __restrict__ wqT,
                           const __bf16* __restrict__ wkT,
                           const __bf16* __restrict__ wvT,
                           __bf16* __restrict__ qb,
                           __bf16* __restrict__ kb,
                           __bf16* __restrict__ vtb) {
  const int wave = blockIdx.x * (blockDim.x >> 5) + (threadIdx.x >> 5);
  const int row0 = wave * 16;
  const int lane = threadIdx.x & 31;
  const int half = lane >> 4, nn = lane & 15;
  const int b = row0 / SEQ;
  const int s0 = row0 % SEQ;

  v16bf a[4];
#pragma unroll
  for (int kc = 0; kc < 4; ++kc) a[kc] = loadA(xb + (size_t)row0 * DIM + kc * 32, DIM);

  const __bf16* Ws[3] = {wqT, wkT, wvT};
#pragma unroll
  for (int t = 0; t < 24; ++t) {          // t = m*8 + n
    const int m = t >> 3, n = t & 7;
    v16bf bt[4];
#pragma unroll
    for (int kc = 0; kc < 4; ++kc)
      bt[kc] = loadB(Ws[m] + (n * 16) * DIM + kc * 32, DIM);
    v8f acc = {0.f, 0.f, 0.f, 0.f, 0.f, 0.f, 0.f, 0.f};
#pragma unroll
    for (int kc = 0; kc < 4; ++kc) acc = wmma_bf16(a[kc], bt[kc], acc);
#pragma unroll
    for (int r = 0; r < 8; ++r) {
      const int M = half * 8 + r;
      const int s = s0 + M;
      const __bf16 val = (__bf16)acc[r];
      if (m == 0)      qb[((b * NH + n) * SEQ + s) * HD + nn] = val;
      else if (m == 1) kb[((b * NH + n) * SEQ + s) * HD + nn] = val;
      else             vtb[((b * NH + n) * HD + nn) * SEQ + s] = val;
    }
  }
}

// ---------------------------------------------------------------------------
// Attention core: one wave per (b, head, 16-query tile); heads 0..6 only.
// Flash-style online softmax, 64 chunks of 32 keys. Pipelined: next chunk's
// K-fragments + async V staging issued before current chunk's softmax.
// ---------------------------------------------------------------------------
__global__ void attn_kernel(const __bf16* __restrict__ qb,
                            const __bf16* __restrict__ kb,
                            const __bf16* __restrict__ vtb,
                            const int* __restrict__ maskv,
                            __bf16* __restrict__ attn) {
  __shared__ __bf16 plds[4][16 * 32];        // P re-striping, 1 KB per wave
#if defined(USE_ASYNC_LDS)
  __shared__ __bf16 vlds[4][2][16 * 32];     // double-buffered V stage, 2 KB/wave
#endif
  const int wslot = threadIdx.x >> 5;
  const int wave = blockIdx.x * (blockDim.x >> 5) + wslot;
  const int stile = wave % (SEQ / 16);
  const int h = (wave / (SEQ / 16)) % 7;
  const int b = wave / ((SEQ / 16) * 7);
  const int lane = threadIdx.x & 31;
  const int half = lane >> 4, nn = lane & 15;
  const int s0 = stile * 16;

  const __bf16* qh = qb + ((size_t)(b * NH + h) * SEQ + s0) * HD;
  const __bf16* kh = kb + (size_t)(b * NH + h) * SEQ * HD;
  const __bf16* vh = vtb + (size_t)(b * NH + h) * HD * SEQ;

  const v16bf qa = loadA_pad16(qh, HD);

  float m[8], l[8];
  v8f out = {0.f, 0.f, 0.f, 0.f, 0.f, 0.f, 0.f, 0.f};
#pragma unroll
  for (int r = 0; r < 8; ++r) { m[r] = -3.0e38f; l[r] = 0.f; }

  // prologue: chunk 0 fragments in flight
  v16bf bk0 = loadB_pad16(kh, HD);
  v16bf bk1 = loadB_pad16(kh + 16 * HD, HD);
#if defined(USE_ASYNC_LDS)
  stage_v(vh, vlds[wslot][0]);
#else
  v16bf bv = loadB(vh, SEQ);
#endif

  const float scale = 0.08838834764831845f;  // 1/sqrt(EMBED_DIM) per reference

  for (int kc = 0; kc < NC; ++kc) {
    const int nk = (kc + 1) & (NC - 1);      // wrap: last iter re-stages chunk 0
    // ---- issue next chunk's staging early (overlaps with softmax below) ----
    v16bf nbk0 = loadB_pad16(kh + (size_t)(nk * 32) * HD, HD);
    v16bf nbk1 = loadB_pad16(kh + (size_t)(nk * 32 + 16) * HD, HD);
#if defined(USE_ASYNC_LDS)
    stage_v(vh + nk * 32, vlds[wslot][nk & 1]);   // 2 async ops -> ASYNCcnt
#endif
#if !defined(USE_ASYNC_LDS)
    v16bf nbv = loadB(vh + nk * 32, SEQ);
#endif
    const int pf = ((kc + 2) & (NC - 1)) * 32;    // warm L2/L0 two chunks ahead
    __builtin_prefetch(kh + (size_t)(pf + nn) * HD, 0, 1);
    __builtin_prefetch(vh + (size_t)nn * SEQ + pf, 0, 1);

    // ---- scores for current chunk ----
    const v8f z = {0.f, 0.f, 0.f, 0.f, 0.f, 0.f, 0.f, 0.f};
    v8f sc0 = wmma_bf16(qa, bk0, z);
    v8f sc1 = wmma_bf16(qa, bk1, z);

    const int t0 = kc * 32 + nn;
    const float madd0 = (maskv[b * SEQ + t0] == 0) ? -1000.f : 0.f;
    const float madd1 = (maskv[b * SEQ + t0 + 16] == 0) ? -1000.f : 0.f;

    float p0[8], p1[8], alpha[8];
#pragma unroll
    for (int r = 0; r < 8; ++r) {
      p0[r] = sc0[r] * scale + madd0;
      p1[r] = sc1[r] * scale + madd1;
    }
    // per-row online softmax (row cols live across 16 lanes of the half-wave)
#pragma unroll
    for (int r = 0; r < 8; ++r) {
      float mx = fmaxf(p0[r], p1[r]);
#pragma unroll
      for (int off = 8; off > 0; off >>= 1) mx = fmaxf(mx, __shfl_xor(mx, off, 32));
      const float mn = fmaxf(m[r], mx);
      alpha[r] = __expf(m[r] - mn);
      p0[r] = __expf(p0[r] - mn);
      p1[r] = __expf(p1[r] - mn);
      float sum = p0[r] + p1[r];
#pragma unroll
      for (int off = 8; off > 0; off >>= 1) sum += __shfl_xor(sum, off, 32);
      l[r] = l[r] * alpha[r] + sum;
      m[r] = mn;
    }
    // C-layout -> A-layout re-striping through LDS (bf16)
    __bf16* pl = plds[wslot];
#pragma unroll
    for (int r = 0; r < 8; ++r) {
      const int M = half * 8 + r;
      pl[M * 32 + nn]      = (__bf16)p0[r];
      pl[M * 32 + 16 + nn] = (__bf16)p1[r];
    }
#pragma unroll
    for (int r = 0; r < 8; ++r) out[r] *= alpha[r];

#if defined(USE_ASYNC_LDS)
    // chunk kc's 2 asyncs done once the 2 just-issued remain outstanding
    __builtin_amdgcn_s_wait_asynccnt(2);
    v16bf bv = ldsB(vlds[wslot][kc & 1]);
#endif
    const v16bf pa = loadA(plds[wslot], 32);
    out = wmma_bf16(pa, bv, out);

    bk0 = nbk0; bk1 = nbk1;
#if !defined(USE_ASYNC_LDS)
    bv = nbv;
#endif
  }

#pragma unroll
  for (int r = 0; r < 8; ++r) out[r] /= l[r];

  // buggy concat: head h -> column block (h<=1 ? h : h+1); head1 also -> 2
  const int jA = (h <= 1) ? h : h + 1;
#pragma unroll
  for (int r = 0; r < 8; ++r) {
    const int M = half * 8 + r;
    const __bf16 val = (__bf16)out[r];
    const size_t rowoff = (size_t)(b * SEQ + s0 + M) * DIM;
    attn[rowoff + jA * 16 + nn] = val;
    if (h == 1) attn[rowoff + 2 * 16 + nn] = val;
  }
}

// ---------------------------------------------------------------------------
// Output projection: attn(bf16) @ lin_w^T + b, f32 out; per-tile B preload.
// ---------------------------------------------------------------------------
__global__ void outproj_kernel(const __bf16* __restrict__ attn,
                               const __bf16* __restrict__ linT,
                               const float* __restrict__ bias,
                               float* __restrict__ out) {
  const int wave = blockIdx.x * (blockDim.x >> 5) + (threadIdx.x >> 5);
  const int row0 = wave * 16;
  const int lane = threadIdx.x & 31;
  const int half = lane >> 4, nn = lane & 15;

  v16bf a[4];
#pragma unroll
  for (int kc = 0; kc < 4; ++kc) a[kc] = loadA(attn + (size_t)row0 * DIM + kc * 32, DIM);

#pragma unroll
  for (int n = 0; n < 8; ++n) {
    v16bf bt[4];
#pragma unroll
    for (int kc = 0; kc < 4; ++kc)
      bt[kc] = loadB(linT + (n * 16) * DIM + kc * 32, DIM);
    v8f acc = {0.f, 0.f, 0.f, 0.f, 0.f, 0.f, 0.f, 0.f};
#pragma unroll
    for (int kc = 0; kc < 4; ++kc) acc = wmma_bf16(a[kc], bt[kc], acc);
    const float bv = bias[n * 16 + nn];
#pragma unroll
    for (int r = 0; r < 8; ++r) {
      const int M = half * 8 + r;
      out[(size_t)(row0 + M) * DIM + n * 16 + nn] = acc[r] + bv;
    }
  }
}

// ---------------------------------------------------------------------------
// Host launcher
// ---------------------------------------------------------------------------
extern "C" void kernel_launch(void* const* d_in, const int* in_sizes, int n_in,
                              void* d_out, int out_size, void* d_ws, size_t ws_size,
                              hipStream_t stream) {
  (void)in_sizes; (void)n_in; (void)out_size; (void)ws_size;
  const float* x   = (const float*)d_in[0];
  const int*  mask = (const int*)d_in[1];
  const float* wq  = (const float*)d_in[2];
  const float* wk  = (const float*)d_in[3];
  const float* wv  = (const float*)d_in[4];
  const float* lw  = (const float*)d_in[5];
  const float* lb  = (const float*)d_in[6];
  float* out = (float*)d_out;

  char* ws = (char*)d_ws;
  size_t off = 0;
  auto carve = [&](size_t bytes) -> char* {
    char* p = ws + off;
    off = (off + bytes + 255) & ~(size_t)255;
    return p;
  };
  __bf16* xb   = (__bf16*)carve((size_t)ROWS * DIM * 2);
  __bf16* wqT  = (__bf16*)carve((size_t)DIM * DIM * 2);
  __bf16* wkT  = (__bf16*)carve((size_t)DIM * DIM * 2);
  __bf16* wvT  = (__bf16*)carve((size_t)DIM * DIM * 2);
  __bf16* linT = (__bf16*)carve((size_t)DIM * DIM * 2);
  __bf16* qb   = (__bf16*)carve((size_t)BSZ * NH * SEQ * HD * 2);
  __bf16* kb   = (__bf16*)carve((size_t)BSZ * NH * SEQ * HD * 2);
  __bf16* vtb  = (__bf16*)carve((size_t)BSZ * NH * HD * SEQ * 2);
  __bf16* attn = (__bf16*)carve((size_t)ROWS * DIM * 2);

  cvt_x_kernel<<<(ROWS * DIM + 255) / 256, 256, 0, stream>>>(x, xb, ROWS * DIM);
  cvt_w_kernel<<<(DIM * DIM + 255) / 256, 256, 0, stream>>>(wq, wk, wv, lw,
                                                            wqT, wkT, wvT, linT);
  qkv_kernel<<<(ROWS / 16) / 4, 128, 0, stream>>>(xb, wqT, wkT, wvT, qb, kb, vtb);
  attn_kernel<<<(BSZ * 7 * (SEQ / 16)) / 4, 128, 0, stream>>>(qb, kb, vtb, mask, attn);
  outproj_kernel<<<(ROWS / 16) / 4, 128, 0, stream>>>(attn, linT, lb, out);
}
